// model3_27814208209094
// MI455X (gfx1250) — matrix operand order repaired and checked by reading, not compiled, not run on hardware
//
#include <hip/hip_runtime.h>

typedef __attribute__((ext_vector_type(16))) _Float16 v16h;
typedef __attribute__((ext_vector_type(8)))  _Float16 v8h;
typedef __attribute__((ext_vector_type(8)))  float    v8f;
typedef __attribute__((ext_vector_type(4)))  float    v4f;

namespace hmm {
constexpr int KH = 16;           // hidden dim per chain
constexpr int S  = 256;          // joint states K*K
constexpr int D  = 88;           // observation dims
constexpr int DP = 96;           // D padded to WMMA K multiple
constexpr int B  = 512;
constexpr int T  = 512;
constexpr int MTILE   = 16;      // batch rows per block
constexpr int NBLOCKS = B / MTILE;
constexpr float NEGF  = -1e30f;

// LDS strides (elements). Row strides are 16B multiples and chosen so the
// 16 lanes of a fragment load hit distinct banks.
constexpr int TM_STR  = 264;     // f16 Tm^T      [S][TM_STR] (row=s', col=s)
constexpr int W_STR   = 104;     // f16 W         [S][W_STR]  (row=s,  col=d)
constexpr int P_STR   = 264;     // f16 exp(a-mx) [16][P_STR]
constexpr int SQF_STR = 100;     // f32 seq stage [16][SQF_STR] (400B rows)
constexpr int SQF_BUF = MTILE * SQF_STR;   // floats per buffer
constexpr int AL_STR  = 264;     // f32 alpha     [16][AL_STR]

constexpr size_t OFF_TM   = 0;
constexpr size_t OFF_W    = OFF_TM   + (size_t)S * TM_STR * 2;      // 135168
constexpr size_t OFF_P    = OFF_W    + (size_t)S * W_STR  * 2;
constexpr size_t OFF_SQF  = OFF_P    + (size_t)MTILE * P_STR * 2;   // 2 bufs
constexpr size_t OFF_AL   = OFF_SQF  + (size_t)2 * SQF_BUF * 4;
constexpr size_t OFF_BIAS = OFF_AL   + (size_t)MTILE * AL_STR * 4;
constexpr size_t OFF_MX   = OFF_BIAS + (size_t)S * 4;
constexpr size_t OFF_LEN  = OFF_MX   + (size_t)16 * 4;
constexpr size_t OFF_PW   = OFF_LEN  + (size_t)16 * 4;
constexpr size_t OFF_PX   = OFF_PW   + (size_t)256 * 4;
constexpr size_t SMEM_BYTES = OFF_PX + (size_t)4096 * 4;            // ~239 KB
} // namespace hmm

__launch_bounds__(256)
__global__ void hmm_forward_fused(const float* __restrict__ seqs,     // [B,T,D]
                                  const int*   __restrict__ lengths,  // [B]
                                  const float* __restrict__ probs_w,  // [K,K]
                                  const float* __restrict__ probs_x,  // [K,K,K]
                                  const float* __restrict__ probs_y,  // [K,K,D]
                                  float*       __restrict__ out)      // [B]
{
  using namespace hmm;
  extern __shared__ char smem[];
  _Float16* TmL  = (_Float16*)(smem + OFF_TM);
  _Float16* WL   = (_Float16*)(smem + OFF_W);
  _Float16* PL   = (_Float16*)(smem + OFF_P);
  float*    SQF  = (float*)(smem + OFF_SQF);
  float*    AL   = (float*)(smem + OFF_AL);
  float*    BIAS = (float*)(smem + OFF_BIAS);
  float*    MX   = (float*)(smem + OFF_MX);
  int*      LEN  = (int*)(smem + OFF_LEN);
  float*    PW   = (float*)(smem + OFF_PW);
  float*    PX   = (float*)(smem + OFF_PX);

  const int tid  = threadIdx.x;
  const int lane = tid & 31;
  const int wv   = tid >> 5;      // wave id 0..7
  const int ln16 = lane & 15;     // A-row / B-col / D-col within tile
  const int hh   = lane >> 4;     // half-wave
  const int b0   = blockIdx.x * MTILE;
  const int rr   = tid >> 4;      // reduction row 0..15
  const int cc   = tid & 15;      // reduction col 0..15

  // Async DMA: stage one observation tile (16 rows x 88 f32 = 352 b128
  // transfers) into LDS. Every wave issues exactly 2 async instructions,
  // so `s_wait_asynccnt 2` == "previous tile complete" for that wave.
  auto issueTile = [&](int t, int buf) {
    const int id0 = wv * 44 + lane;                 // transfers 0..351
    {
      const float* g = seqs + ((size_t)(b0 + id0 / 22) * T + t) * D + (id0 % 22) * 4;
      const unsigned l = (unsigned)(size_t)(const void*)
          (SQF + buf * SQF_BUF + (id0 / 22) * SQF_STR + (id0 % 22) * 4);
      asm volatile("global_load_async_to_lds_b128 %0, %1, off"
                   :: "v"(l), "v"((unsigned long long)(size_t)g) : "memory");
    }
    if (lane < 12) {                                // remaining 12 per wave
      const int id1 = id0 + 32;
      const float* g = seqs + ((size_t)(b0 + id1 / 22) * T + t) * D + (id1 % 22) * 4;
      const unsigned l = (unsigned)(size_t)(const void*)
          (SQF + buf * SQF_BUF + (id1 / 22) * SQF_STR + (id1 % 22) * 4);
      asm volatile("global_load_async_to_lds_b128 %0, %1, off"
                   :: "v"(l), "v"((unsigned long long)(size_t)g) : "memory");
    }
  };

  // Kick off tile 0 immediately: the DMA overlaps the whole setup phase.
  issueTile(0, 0);

  // ---- stage parameters into LDS ----
  PW[tid] = probs_w[tid];
  for (int i = tid; i < KH * KH * KH; i += 256) PX[i] = probs_x[i];
  if (tid < MTILE) LEN[tid] = lengths[b0 + tid];
  // zero pad columns 88..99 of both staging buffers (DMA never writes them)
  for (int i = tid; i < 2 * MTILE * 12; i += 256) {
    const int br = i / 12, d = D + i % 12;
    SQF[(br / MTILE) * SQF_BUF + (br % MTILE) * SQF_STR + d] = 0.f;
  }
  __syncthreads();

  // Joint transition, stored transposed for B-fragment reads:
  // TmL[s'][s] = pw[w,w'] * px[w', x, x'],  s=(w,x), s'=(w',x')
  for (int i = tid; i < S * S; i += 256) {
    const int sp = i >> 8, s = i & (S - 1);
    const int w = s >> 4, x = s & 15, wp = sp >> 4, xp = sp & 15;
    TmL[sp * TM_STR + s] = (_Float16)(PW[w * KH + wp] * PX[(wp * KH + x) * KH + xp]);
  }
  // Emission weights WL[s][d] = log(p)-log1p(-p), BIAS[s] = sum_d log1p(-p)
  {
    const int s = tid;
    float bsum = 0.f;
    for (int d = 0; d < D; ++d) {
      const float p   = probs_y[s * D + d];
      const float l1m = log1pf(-p);
      WL[s * W_STR + d] = (_Float16)(__logf(p) - l1m);
      bsum += l1m;
    }
    for (int d = D; d < DP; ++d) WL[s * W_STR + d] = (_Float16)0.f;
    BIAS[s] = bsum;
  }
  // alpha0: joint state 0 -> 0, else -1e30
  for (int i = tid; i < MTILE * S; i += 256) {
    const int r = i >> 8, s = i & (S - 1);
    AL[r * AL_STR + s] = (s == 0) ? 0.f : NEGF;
  }
  __syncthreads();

  int tmax = 1;
  #pragma unroll
  for (int r = 0; r < MTILE; ++r) { const int l = LEN[r]; tmax = l > tmax ? l : tmax; }

  // width-16 shuffle reductions (row groups are contiguous lanes in a wave)
  auto shflMax16 = [&](float v) {
    #pragma unroll
    for (int m = 8; m >= 1; m >>= 1) v = fmaxf(v, __shfl_xor(v, m, 16));
    return v;
  };
  auto shflSum16 = [&](float v) {
    #pragma unroll
    for (int m = 8; m >= 1; m >>= 1) v += __shfl_xor(v, m, 16);
    return v;
  };

  // recompute per-row max + P = exp(alpha - max) (f16). No internal barrier;
  // callers rely on the loop-top / final barriers.
  auto refresh = [&]() {
    float lmax = NEGF;
    #pragma unroll
    for (int i = 0; i < 16; ++i)
      lmax = fmaxf(lmax, AL[rr * AL_STR + cc * 16 + i]);
    const float m = shflMax16(lmax);
    if (cc == 0) MX[rr] = m;
    #pragma unroll
    for (int i = 0; i < 16; ++i) {
      const int s = cc * 16 + i;
      PL[rr * P_STR + s] = (_Float16)__expf(AL[rr * AL_STR + s] - m);
    }
  };
  refresh();

  // A fragment (16x32 f16): two ds_load_b128 per lane
  auto ldA = [&](const _Float16* base, int stride, int kt) {
    const _Float16* p = base + ln16 * stride + kt * 32 + hh * 8;
    v8h lo = *(const v8h*)(p);
    v8h hi = *(const v8h*)(p + 16);
    v16h r = __builtin_shufflevector(lo, hi, 0,1,2,3,4,5,6,7,8,9,10,11,12,13,14,15);
    return r;
  };
  // A fragment from f32 staging: four ds_load_b128 + pack to f16
  auto ldA32 = [&](const float* base, int kt) {
    const float* p = base + ln16 * SQF_STR + kt * 32 + hh * 8;
    v4f a0 = *(const v4f*)(p);
    v4f a1 = *(const v4f*)(p + 4);
    v4f a2 = *(const v4f*)(p + 16);
    v4f a3 = *(const v4f*)(p + 20);
    v16h r;
    #pragma unroll
    for (int e = 0; e < 4; ++e) {
      r[e]      = (_Float16)a0[e];
      r[e + 4]  = (_Float16)a1[e];
      r[e + 8]  = (_Float16)a2[e];
      r[e + 12] = (_Float16)a3[e];
    }
    return r;
  };
  // B fragment (32x16 f16), source stored [N][K]: two ds_load_b128 per lane
  auto ldB = [&](const _Float16* base, int stride, int nt, int kt) {
    const _Float16* p = base + (nt * 16 + ln16) * stride + kt * 32 + hh * 8;
    v8h lo = *(const v8h*)(p);
    v8h hi = *(const v8h*)(p + 16);
    v16h r = __builtin_shufflevector(lo, hi, 0,1,2,3,4,5,6,7,8,9,10,11,12,13,14,15);
    return r;
  };

  const int nt0 = wv * 2, nt1 = nt0 + 1;   // 2 N-tiles per wave

  // hoist loop invariants: emission B fragments + per-row lengths
  v16h wb0[3], wb1[3];
  #pragma unroll
  for (int kt = 0; kt < 3; ++kt) {
    wb0[kt] = ldB(WL, W_STR, nt0, kt);
    wb1[kt] = ldB(WL, W_STR, nt1, kt);
  }
  int lenR[8];
  #pragma unroll
  for (int j = 0; j < 8; ++j) lenR[j] = LEN[j + 8 * hh];

  int cur = 0;
  for (int t = 0; t < tmax; ++t, cur ^= 1) {
    // double-buffered async staging: issue t+1, then wait for tile t
    if (t + 1 < tmax) {
      issueTile(t + 1, cur ^ 1);
      asm volatile("s_wait_asynccnt 0x2" ::: "memory");
    } else {
      asm volatile("s_wait_asynccnt 0x0" ::: "memory");
    }
    __syncthreads();   // all waves' DMA for tile t visible; prev P/MX stable

    v8f aT0 = {}; v8f aT1 = {}; v8f aE0 = {}; v8f aE1 = {};
    // transition: [16 x 256] @ [256 x 256], K in 8 steps of 32
    #pragma unroll
    for (int kt = 0; kt < 8; ++kt) {
      v16h a   = ldA(PL, P_STR, kt);
      v16h bt0 = ldB(TmL, TM_STR, nt0, kt);
      v16h bt1 = ldB(TmL, TM_STR, nt1, kt);
      aT0 = __builtin_amdgcn_wmma_f32_16x16x32_f16(false, a, false, bt0, (short)0, aT0, false, false);
      aT1 = __builtin_amdgcn_wmma_f32_16x16x32_f16(false, a, false, bt1, (short)0, aT1, false, false);
    }
    // emission: [16 x 96] @ [96 x 256], K in 3 steps of 32 (A from f32 stage)
    #pragma unroll
    for (int kt = 0; kt < 3; ++kt) {
      v16h a = ldA32(SQF + cur * SQF_BUF, kt);
      aE0 = __builtin_amdgcn_wmma_f32_16x16x32_f16(false, a, false, wb0[kt], (short)0, aE0, false, false);
      aE1 = __builtin_amdgcn_wmma_f32_16x16x32_f16(false, a, false, wb1[kt], (short)0, aE1, false, false);
    }

    // alpha' = mx + log(P@Tm) + (seq@W + bias), masked by per-row length.
    // D layout: VGPR j -> (M = j + 8*hh, N = ntile*16 + ln16)
    #pragma unroll
    for (int j = 0; j < 8; ++j) {
      const int m = j + 8 * hh;
      if (t < lenR[j]) {
        const int n0 = nt0 * 16 + ln16;
        const int n1 = nt1 * 16 + ln16;
        const float base = MX[m];
        AL[m * AL_STR + n0] = base + __logf(aT0[j]) + aE0[j] + BIAS[n0];
        AL[m * AL_STR + n1] = base + __logf(aT1[j]) + aE1[j] + BIAS[n1];
      }
    }
    __syncthreads();   // alpha stable before refresh reads
    refresh();
  }
  __syncthreads();

  // final per-row logsumexp (shuffle reduction within the row's 16 lanes)
  {
    const float m = MX[rr];
    float psum = 0.f;
    #pragma unroll
    for (int i = 0; i < 16; ++i)
      psum += __expf(AL[rr * AL_STR + cc * 16 + i] - m);
    const float tot = shflSum16(psum);
    if (cc == 0) out[b0 + rr] = m + __logf(tot);
  }
}

extern "C" void kernel_launch(void* const* d_in, const int* in_sizes, int n_in,
                              void* d_out, int out_size, void* d_ws, size_t ws_size,
                              hipStream_t stream) {
  (void)in_sizes; (void)n_in; (void)out_size; (void)d_ws; (void)ws_size;
  const float* seqs    = (const float*)d_in[0];
  const int*   lengths = (const int*)d_in[1];
  const float* pw      = (const float*)d_in[2];
  const float* px      = (const float*)d_in[3];
  const float* py      = (const float*)d_in[4];
  float*       out     = (float*)d_out;

  // WGP-mode LDS: 320 KB per workgroup; we use ~239 KB dynamically.
  (void)hipFuncSetAttribute((const void*)hmm_forward_fused,
                            hipFuncAttributeMaxDynamicSharedMemorySize,
                            (int)hmm::SMEM_BYTES);
  hmm_forward_fused<<<dim3(hmm::NBLOCKS), dim3(256), hmm::SMEM_BYTES, stream>>>(
      seqs, lengths, pw, px, py, out);
}
